// LSTMAttention2_1297080123473
// MI455X (gfx1250) — compile-verified, matrix operand order checked by reading, not run on hardware
//
#include <hip/hip_runtime.h>
#include <hip/hip_bf16.h>
#include <cmath>

#define B_  128
#define S_  256
#define D_  64
#define H_  512
#define H2_ 1024
#define GE_ 2048
#define GD_ 4096
#define T_  60
#define KC_ 512   // K-chunk staged in LDS (halves)

typedef _Float16 f16;
typedef __attribute__((ext_vector_type(16))) _Float16 v16h;
typedef __attribute__((ext_vector_type(8)))  _Float16 v8h;
typedef __attribute__((ext_vector_type(8)))  float    v8f;

union frag16 { v16h v; v8h h[2]; };

// ---- CDNA5 async global->LDS copy (16 bytes per lane), ASYNCcnt-tracked ----
__device__ inline void async_ld16(unsigned lds_byte_addr, const f16* g) {
  asm volatile("global_load_async_to_lds_b128 %0, %1, off"
               :: "v"(lds_byte_addr), "v"((unsigned long long)(size_t)g)
               : "memory");
}
__device__ inline void wait_async0() {
#if __has_builtin(__builtin_amdgcn_s_wait_asynccnt)
  __builtin_amdgcn_s_wait_asynccnt(0);
#else
  asm volatile("s_wait_asynccnt 0x0" ::: "memory");
#endif
}

// Load one 16x32 f16 B fragment from W stored [N,K] row-major (acts as B^T).
// ISA layout: lanes 0-15 rows 0-15 / K chunks {0..7,16..23}, lanes 16-31 {8..15,24..31}.
__device__ inline v16h load_fragW(const f16* __restrict__ base, int ld, int row0, int k0, int lane) {
  const f16* p = base + (size_t)(row0 + (lane & 15)) * (size_t)ld
                      + (size_t)(k0 + ((lane >> 4) << 3));
  frag16 u;
  u.h[0] = *(const v8h*)(p);
  u.h[1] = *(const v8h*)(p + 16);
  __builtin_prefetch(p + 32, 0, 1);   // global_prefetch for next K chunk
  return u.v;
}

// C[M,N] = A[M,K] @ W[N,K]^T (+bias[N]).
// Block = 8 waves, one 16-row m-tile; A panel staged to LDS via async copy and
// shared by all waves; each wave computes a 16x64 strip (4 WMMA tiles).
template <typename OutT>
__global__ void gemm_wmma(const f16* __restrict__ A, const f16* __restrict__ W,
                          const float* __restrict__ bias, OutT* __restrict__ C,
                          int M, int N, int K, int lda, int ldw, int ldc) {
  __shared__ f16 sA[16 * KC_];
  int nStrips = (N + 511) >> 9;                 // 512 cols per block
  int mTile = blockIdx.x / nStrips;
  int strip = blockIdx.x - mTile * nStrips;
  int m0 = mTile << 4;
  int tid  = threadIdx.x;
  int wave = tid >> 5;
  int lane = tid & 31;
  int n0 = (strip << 9) + (wave << 6);          // this wave's 64-col strip
  bool active = (n0 < N);                       // wave-uniform (N % 64 == 0)
  unsigned sbase = (unsigned)(size_t)(void*)sA; // low 32 bits = LDS offset

  v8f acc0 = {}, acc1 = {}, acc2 = {}, acc3 = {};

  for (int k0 = 0; k0 < K; k0 += KC_) {
    int kc = K - k0; if (kc > KC_) kc = KC_;    // 512 or ragged 64 tail
    // ---- stage A[m0:m0+16, k0:k0+kc] into LDS (async, all 8 waves help) ----
    int cpr = kc >> 3;                          // 16B chunks per row
    int total = cpr << 4;                       // 16 rows
    for (int c = tid; c < total; c += 256) {
      int r   = c / cpr;
      int col = c - r * cpr;
      const f16* g = A + (size_t)(m0 + r) * (size_t)lda + (size_t)(k0 + (col << 3));
      async_ld16(sbase + (unsigned)((r * KC_ + (col << 3)) * 2), g);
    }
    wait_async0();
    __syncthreads();
    // ---- compute: A fragments from LDS, B fragments streamed from global ----
    if (active) {
      for (int kk = 0; kk < kc; kk += 32) {
        const f16* pa = sA + (size_t)(lane & 15) * KC_ + (size_t)(kk + ((lane >> 4) << 3));
        frag16 ua;
        ua.h[0] = *(const v8h*)(pa);            // ds_load_b128
        ua.h[1] = *(const v8h*)(pa + 16);
        v16h a  = ua.v;
        v16h b0 = load_fragW(W, ldw, n0,      k0 + kk, lane);
        v16h b1 = load_fragW(W, ldw, n0 + 16, k0 + kk, lane);
        v16h b2 = load_fragW(W, ldw, n0 + 32, k0 + kk, lane);
        v16h b3 = load_fragW(W, ldw, n0 + 48, k0 + kk, lane);
        acc0 = __builtin_amdgcn_wmma_f32_16x16x32_f16(false, a, false, b0, (short)0, acc0, false, false);
        acc1 = __builtin_amdgcn_wmma_f32_16x16x32_f16(false, a, false, b1, (short)0, acc1, false, false);
        acc2 = __builtin_amdgcn_wmma_f32_16x16x32_f16(false, a, false, b2, (short)0, acc2, false, false);
        acc3 = __builtin_amdgcn_wmma_f32_16x16x32_f16(false, a, false, b3, (short)0, acc3, false, false);
      }
    }
    __syncthreads();
  }

  if (!active) return;
  int rb   = m0 + ((lane >> 4) << 3);           // lanes 16-31 handle rows +8
  int colL = lane & 15;
#pragma unroll
  for (int j = 0; j < 4; ++j) {
    int col = n0 + (j << 4) + colL;
    float bv = bias ? bias[col] : 0.0f;
    v8f acc = (j == 0) ? acc0 : (j == 1) ? acc1 : (j == 2) ? acc2 : acc3;
#pragma unroll
    for (int r = 0; r < 8; ++r)
      C[(size_t)(rb + r) * (size_t)ldc + (size_t)col] = (OutT)(acc[r] + bv);
  }
}

// Concatenate up to 3 (rows x k_i) panels (each f32 or f16) into a packed f16 panel.
__global__ void pack3(f16* __restrict__ out, int rows, int Ktot,
                      const void* s0, int k0, int ld0, int h0,
                      const void* s1, int k1, int ld1, int h1,
                      const void* s2, int k2, int ld2, int h2) {
  int idx = blockIdx.x * blockDim.x + threadIdx.x;
  if (idx >= rows * Ktot) return;
  int r = idx / Ktot;
  int k = idx - r * Ktot;
  const void* s; int ld, ih, kk;
  if (k < k0)           { s = s0; ld = ld0; ih = h0; kk = k; }
  else if (k < k0 + k1) { s = s1; ld = ld1; ih = h1; kk = k - k0; }
  else                  { s = s2; ld = ld2; ih = h2; kk = k - k0 - k1; }
  float v = ih ? (float)((const f16*)s)[(size_t)r * (size_t)ld + kk]
               :        ((const float*)s)[(size_t)r * (size_t)ld + kk];
  out[(size_t)r * (size_t)Ktot + k] = (f16)v;
}

__device__ inline float sigm(float x) { return 1.0f / (1.0f + __expf(-x)); }

// Fused bias + LSTM gate activations.  gates: (B, 4*Hd) pre-activation (no bias).
__global__ void lstm_gates(const float* __restrict__ gates, const float* __restrict__ bias,
                           int Hd, float* __restrict__ h, float* __restrict__ c,
                           f16* __restrict__ out16, int out16_ld) {
  int idx = blockIdx.x * blockDim.x + threadIdx.x;
  if (idx >= B_ * Hd) return;
  int b = idx / Hd;
  int j = idx - b * Hd;
  const float* g = gates + (size_t)b * (size_t)(4 * Hd);
  float gi = sigm (g[j]          + bias[j]);
  float gf = sigm (g[Hd + j]     + bias[Hd + j]);
  float gg = tanhf(g[2 * Hd + j] + bias[2 * Hd + j]);
  float go = sigm (g[3 * Hd + j] + bias[3 * Hd + j]);
  float cc = gf * c[idx] + gi * gg;
  c[idx] = cc;
  float hh = go * tanhf(cc);
  h[idx] = hh;
  if (out16) out16[(size_t)b * (size_t)out16_ld + j] = (f16)hh;
}

// One block per batch row: scores -> softmax -> context, all in LDS.
__global__ void attn_kernel(const float* __restrict__ h, const f16* __restrict__ keys,
                            const f16* __restrict__ enc, float* __restrict__ ctx) {
  __shared__ float hs[H2_];
  __shared__ float sc[S_];
  __shared__ float red[S_];
  int b = blockIdx.x;
  int t = threadIdx.x;                       // 256 threads == S_
  for (int k = t; k < H2_; k += S_) hs[k] = h[(size_t)b * H2_ + k];
  __syncthreads();
  const f16* krow = keys + ((size_t)b * S_ + t) * H2_;
  float s = 0.0f;
  for (int k = 0; k < H2_; k += 8) {
    v8h kv = *(const v8h*)(krow + k);
#pragma unroll
    for (int j = 0; j < 8; ++j) s += hs[k + j] * (float)kv[j];
  }
  red[t] = s;
  __syncthreads();
  for (int off = S_ / 2; off > 0; off >>= 1) {
    if (t < off) red[t] = fmaxf(red[t], red[t + off]);
    __syncthreads();
  }
  float mx = red[0];
  __syncthreads();
  float e = __expf(s - mx);
  sc[t]  = e;
  red[t] = e;
  __syncthreads();
  for (int off = S_ / 2; off > 0; off >>= 1) {
    if (t < off) red[t] += red[t + off];
    __syncthreads();
  }
  float inv = 1.0f / red[0];
  __syncthreads();
  for (int d = t; d < H2_; d += S_) {
    const f16* ep = enc + (size_t)b * S_ * H2_ + d;
    float acc = 0.0f;
    for (int s2 = 0; s2 < S_; ++s2) acc += sc[s2] * (float)ep[(size_t)s2 * H2_];
    ctx[(size_t)b * H2_ + d] = acc * inv;
  }
}

extern "C" void kernel_launch(void* const* d_in, const int* in_sizes, int n_in,
                              void* d_out, int out_size, void* d_ws, size_t ws_size,
                              hipStream_t stream) {
  const float* x        = (const float*)d_in[0];
  const float* enc0_Wih = (const float*)d_in[1];
  const float* enc0_Whh = (const float*)d_in[2];
  const float* enc0_b   = (const float*)d_in[3];
  const float* enc1_Wih = (const float*)d_in[4];
  const float* enc1_Whh = (const float*)d_in[5];
  const float* enc1_b   = (const float*)d_in[6];
  const float* attnW[3] = {(const float*)d_in[7], (const float*)d_in[9],  (const float*)d_in[11]};
  const float* attnB[3] = {(const float*)d_in[8], (const float*)d_in[10], (const float*)d_in[12]};
  const float* dec_Wih[3] = {(const float*)d_in[13], (const float*)d_in[16], (const float*)d_in[19]};
  const float* dec_Whh[3] = {(const float*)d_in[14], (const float*)d_in[17], (const float*)d_in[20]};
  const float* dec_b[3]   = {(const float*)d_in[15], (const float*)d_in[18], (const float*)d_in[21]};
  const float* out_W = (const float*)d_in[22];
  const float* out_b = (const float*)d_in[23];
  float* out = (float*)d_out;

  // ---- workspace carve ----
  char* ws = (char*)d_ws;
  size_t off = 0;
  auto alloc = [&](size_t bytes) {
    void* p = ws + off;
    off = (off + bytes + 255) & ~(size_t)255;
    return p;
  };
  const int KE0 = D_ + H_;          // 576
  const int KE1 = H2_ + H_;         // 1536
  const int KD1 = H2_ + D_ + H2_;   // 2112
  const int KD23 = 3 * H2_;         // 3072
  f16* wEnc0[2]; f16* wEnc1[2]; f16* wAttn[3]; f16* keysB[3];
  for (int d = 0; d < 2; ++d) wEnc0[d] = (f16*)alloc((size_t)GE_ * KE0 * 2);
  for (int d = 0; d < 2; ++d) wEnc1[d] = (f16*)alloc((size_t)GE_ * KE1 * 2);
  f16* wDec1 = (f16*)alloc((size_t)GD_ * KD1 * 2);
  f16* wDec2 = (f16*)alloc((size_t)GD_ * KD23 * 2);
  f16* wDec3 = (f16*)alloc((size_t)GD_ * KD23 * 2);
  for (int k = 0; k < 3; ++k) wAttn[k] = (f16*)alloc((size_t)H2_ * H2_ * 2);
  f16* wOut = (f16*)alloc((size_t)D_ * H2_ * 2);
  f16* enc0_seq = (f16*)alloc((size_t)B_ * S_ * H2_ * 2);
  f16* enc_seq  = (f16*)alloc((size_t)B_ * S_ * H2_ * 2);
  for (int k = 0; k < 3; ++k) keysB[k] = (f16*)alloc((size_t)B_ * S_ * H2_ * 2);
  float* gates = (float*)alloc((size_t)B_ * GD_ * 4);
  f16*   packA = (f16*)alloc((size_t)B_ * KD23 * 2);
  f16*   h3h   = (f16*)alloc((size_t)B_ * H2_ * 2);
  float* hE = (float*)alloc((size_t)B_ * H_ * 4);
  float* cE = (float*)alloc((size_t)B_ * H_ * 4);
  float* h1 = (float*)alloc((size_t)B_ * H2_ * 4);
  float* c1 = (float*)alloc((size_t)B_ * H2_ * 4);
  float* h2 = (float*)alloc((size_t)B_ * H2_ * 4);
  float* c2 = (float*)alloc((size_t)B_ * H2_ * 4);
  float* h3 = (float*)alloc((size_t)B_ * H2_ * 4);
  float* c3 = (float*)alloc((size_t)B_ * H2_ * 4);
  float* ctx = (float*)alloc((size_t)B_ * H2_ * 4);

  auto gemmF = [&](const f16* A, const f16* W, const float* bias, float* C,
                   int M, int N, int K, int lda, int ldw, int ldc) {
    int nStrips = (N + 511) / 512;
    gemm_wmma<float><<<(M >> 4) * nStrips, 256, 0, stream>>>(A, W, bias, C, M, N, K, lda, ldw, ldc);
  };
  auto gemmH = [&](const f16* A, const f16* W, const float* bias, f16* C,
                   int M, int N, int K, int lda, int ldw, int ldc) {
    int nStrips = (N + 511) / 512;
    gemm_wmma<f16><<<(M >> 4) * nStrips, 256, 0, stream>>>(A, W, bias, C, M, N, K, lda, ldw, ldc);
  };
  auto pack = [&](f16* o, int rows, int Ktot,
                  const void* s0, int k0, int ld0, int ih0,
                  const void* s1, int k1, int ld1, int ih1,
                  const void* s2, int k2, int ld2, int ih2) {
    int n = rows * Ktot;
    pack3<<<(n + 255) / 256, 256, 0, stream>>>(o, rows, Ktot,
        s0, k0, ld0, ih0, s1, k1, ld1, ih1, s2, k2, ld2, ih2);
  };
  auto gatesK = [&](const float* g, const float* bias, int Hd, float* h, float* c,
                    f16* o16, int o16ld) {
    int n = B_ * Hd;
    lstm_gates<<<(n + 255) / 256, 256, 0, stream>>>(g, bias, Hd, h, c, o16, o16ld);
  };

  // ---- per-call weight prep: fuse [Wih | Whh] and convert to f16 ----
  for (int d = 0; d < 2; ++d) {
    pack(wEnc0[d], GE_, KE0, enc0_Wih + (size_t)d * GE_ * D_,  D_,  D_,  0,
                             enc0_Whh + (size_t)d * GE_ * H_,  H_,  H_,  0, nullptr, 0, 0, 0);
    pack(wEnc1[d], GE_, KE1, enc1_Wih + (size_t)d * GE_ * H2_, H2_, H2_, 0,
                             enc1_Whh + (size_t)d * GE_ * H_,  H_,  H_,  0, nullptr, 0, 0, 0);
  }
  pack(wDec1, GD_, KD1,  dec_Wih[0], H2_ + D_,  H2_ + D_,  0, dec_Whh[0], H2_, H2_, 0, nullptr, 0, 0, 0);
  pack(wDec2, GD_, KD23, dec_Wih[1], 2 * H2_,   2 * H2_,   0, dec_Whh[1], H2_, H2_, 0, nullptr, 0, 0, 0);
  pack(wDec3, GD_, KD23, dec_Wih[2], 2 * H2_,   2 * H2_,   0, dec_Whh[2], H2_, H2_, 0, nullptr, 0, 0, 0);
  for (int k = 0; k < 3; ++k)
    pack(wAttn[k], H2_, H2_, attnW[k], H2_, H2_, 0, nullptr, 0, 0, 0, nullptr, 0, 0, 0);
  pack(wOut, D_, H2_, out_W, H2_, H2_, 0, nullptr, 0, 0, 0, nullptr, 0, 0, 0);

  // ---- encoder: 2 layers x 2 directions, sequential scan ----
  for (int layer = 0; layer < 2; ++layer) {
    for (int d = 0; d < 2; ++d) {
      hipMemsetAsync(hE, 0, (size_t)B_ * H_ * 4, stream);
      hipMemsetAsync(cE, 0, (size_t)B_ * H_ * 4, stream);
      int Kt = (layer == 0) ? KE0 : KE1;
      const f16* Wf = (layer == 0) ? wEnc0[d] : wEnc1[d];
      const float* bptr = ((layer == 0) ? enc0_b : enc1_b) + (size_t)d * GE_;
      f16* seq = (layer == 0) ? enc0_seq : enc_seq;
      for (int st = 0; st < S_; ++st) {
        int t = (d == 0) ? st : (S_ - 1 - st);
        if (layer == 0)
          pack(packA, B_, Kt, x + (size_t)t * D_, D_, S_ * D_, 0,
               hE, H_, H_, 0, nullptr, 0, 0, 0);
        else
          pack(packA, B_, Kt, enc0_seq + (size_t)t * H2_, H2_, S_ * H2_, 1,
               hE, H_, H_, 0, nullptr, 0, 0, 0);
        gemmF(packA, Wf, nullptr, gates, B_, GE_, Kt, Kt, Kt, GE_);
        gatesK(gates, bptr, H_, hE, cE, seq + (size_t)t * H2_ + (size_t)d * H_, S_ * H2_);
      }
    }
  }

  // ---- attention keys (precompute once, store f16) ----
  for (int k = 0; k < 3; ++k)
    gemmH(enc_seq, wAttn[k], attnB[k], keysB[k], B_ * S_, H2_, H2_, H2_, H2_, H2_);

  // ---- decoder ----
  hipMemsetAsync(h1, 0, (size_t)B_ * H2_ * 4, stream);
  hipMemsetAsync(c1, 0, (size_t)B_ * H2_ * 4, stream);
  hipMemsetAsync(h2, 0, (size_t)B_ * H2_ * 4, stream);
  hipMemsetAsync(c2, 0, (size_t)B_ * H2_ * 4, stream);
  hipMemsetAsync(h3, 0, (size_t)B_ * H2_ * 4, stream);
  hipMemsetAsync(c3, 0, (size_t)B_ * H2_ * 4, stream);

  for (int t = 0; t < T_; ++t) {
    const float* prevP = (t == 0) ? (x + (size_t)(S_ - 1) * D_) : (out + (size_t)(t - 1) * D_);
    int prevLd = (t == 0) ? (S_ * D_) : (T_ * D_);
    // cell 1
    attn_kernel<<<B_, S_, 0, stream>>>(h1, keysB[0], enc_seq, ctx);
    pack(packA, B_, KD1, ctx, H2_, H2_, 0, prevP, D_, prevLd, 0, h1, H2_, H2_, 0);
    gemmF(packA, wDec1, nullptr, gates, B_, GD_, KD1, KD1, KD1, GD_);
    gatesK(gates, dec_b[0], H2_, h1, c1, nullptr, 0);
    // cell 2
    attn_kernel<<<B_, S_, 0, stream>>>(h2, keysB[1], enc_seq, ctx);
    pack(packA, B_, KD23, ctx, H2_, H2_, 0, h1, H2_, H2_, 0, h2, H2_, H2_, 0);
    gemmF(packA, wDec2, nullptr, gates, B_, GD_, KD23, KD23, KD23, GD_);
    gatesK(gates, dec_b[1], H2_, h2, c2, nullptr, 0);
    // cell 3
    attn_kernel<<<B_, S_, 0, stream>>>(h3, keysB[2], enc_seq, ctx);
    pack(packA, B_, KD23, ctx, H2_, H2_, 0, h2, H2_, H2_, 0, h3, H2_, H2_, 0);
    gemmF(packA, wDec3, nullptr, gates, B_, GD_, KD23, KD23, KD23, GD_);
    gatesK(gates, dec_b[2], H2_, h3, c3, nullptr, 0);
    // output projection -> d_out[:, t, :]  (ldc = T*D scatters into (B,T,D))
    pack(h3h, B_, H2_, h3, H2_, H2_, 0, nullptr, 0, 0, 0, nullptr, 0, 0, 0);
    gemmF(h3h, wOut, out_b, out + (size_t)t * D_, B_, D_, H2_, H2_, H2_, T_ * D_);
  }
}
